// MiM_v2_91070486544638
// MI455X (gfx1250) — matrix operand order repaired and checked by reading, not compiled, not run on hardware
//
#include <hip/hip_runtime.h>
#include <hip/hip_bf16.h>
#include <math.h>

// ---------------------------------------------------------------------------
// Mamba block for gfx1250 (MI455X): bf16 WMMA GEMMs + f32 selective scan.
// GEMM waves are register-blocked 32x64 (2x4 WMMA tiles) for fragment reuse.
// ---------------------------------------------------------------------------

typedef __attribute__((ext_vector_type(16))) __bf16 v16bf;
typedef __attribute__((ext_vector_type(8)))  float  v8f;
typedef int v4i __attribute__((vector_size(16)));   // matches async-LDS builtin

#define Bb   16
#define Ll   1024
#define DM   512
#define DIN  1024
#define DT   32
#define Ns   16
#define Mrows (Bb * Ll)    // 16384
#define DBC  (DT + 2 * Ns) // 64

#if __has_builtin(__builtin_amdgcn_global_load_async_to_lds_b128) && \
    __has_builtin(__builtin_amdgcn_s_wait_asynccnt)
#define USE_ASYNC_LDS 1
#else
#define USE_ASYNC_LDS 0
#endif

// ---- force global (AS1) memory ops so codegen emits global_load/global_store
// ---- (not flat_*, which also ties up the LDS path / DScnt on CDNA5)
__device__ __forceinline__ v16bf gload_v16bf(const __bf16* p) {
    return *(const __attribute__((address_space(1))) v16bf*)p;
}
__device__ __forceinline__ float gload_f32(const float* p) {
    return *(const __attribute__((address_space(1))) float*)p;
}
__device__ __forceinline__ void gstore_f32(float* p, float v) {
    *(__attribute__((address_space(1))) float*)p = v;
}
__device__ __forceinline__ void gstore_bf16(__bf16* p, __bf16 v) {
    *(__attribute__((address_space(1))) __bf16*)p = v;
}

// --------------------------- fragment packing ------------------------------
// A-matrix (16x32 bf16) per-lane layout (ISA 7.12.2):
//   lanes 0-15  : m = lane,    elems 0..7 -> K 0..7,   elems 8..15 -> K 16..23
//   lanes 16-31 : m = lane-16, elems 0..7 -> K 8..15,  elems 8..15 -> K 24..31
__device__ __forceinline__ size_t a_pack_idx(int m, int k, int ksteps) {
    int mt = m >> 4, r = m & 15;
    int ks = k >> 5, kk = k & 31;
    int lane, j;
    if      (kk <  8) { lane = r;      j = kk;      }
    else if (kk < 16) { lane = r + 16; j = kk - 8;  }
    else if (kk < 24) { lane = r;      j = kk - 8;  }
    else              { lane = r + 16; j = kk - 16; }
    return ((size_t)(mt * ksteps + ks) * 32 + lane) * 16 + j;
}

// B-matrix (32x16 bf16): lanes 0-15 hold K 0..7 (elems 0..7) and K 16..23
// (elems 8..15), lanes 16-31 hold K 8..15 and K 24..31, n = lane & 15.
__device__ __forceinline__ size_t b_pack_idx(int k, int n, int ksteps) {
    int nt = n >> 4, c = n & 15;
    int ks = k >> 5, kk = k & 31;
    int lane = c + (((kk >> 3) & 1) << 4);
    int j = (kk & 7) + ((kk >> 4) << 3);
    return ((size_t)(nt * ksteps + ks) * 32 + lane) * 16 + j;
}

// --------------------------- prep kernels ----------------------------------
__global__ void pack_a_kernel(const float* __restrict__ src, __bf16* __restrict__ dst,
                              int kdim) {
    int idx = blockIdx.x * 256 + threadIdx.x;
    int m = idx / kdim, k = idx - m * kdim;
    gstore_bf16(&dst[a_pack_idx(m, k, kdim >> 5)], (__bf16)gload_f32(&src[idx]));
}

__global__ void pack_b_kernel(const float* __restrict__ w, __bf16* __restrict__ dst,
                              int kdim) {
    int idx = blockIdx.x * 256 + threadIdx.x;
    int n = idx / kdim, k = idx - n * kdim;
    gstore_bf16(&dst[b_pack_idx(k, n, kdim >> 5)], (__bf16)gload_f32(&w[idx]));
}

__global__ void aneg_kernel(const float* __restrict__ alog, float* __restrict__ aneg) {
    int i = blockIdx.x * 256 + threadIdx.x;
    gstore_f32(&aneg[i], -__expf(gload_f32(&alog[i])));
}

// --------------------------- WMMA GEMM -------------------------------------
// C[M, nout] = A[M,K] * B[K, nout] (+bias), bf16 inputs, f32 accum.
// One wave computes a 32x64 block = 2x4 WMMA tiles (8 accumulators), so each
// k-step issues 8 v_wmma per 12 b128 loads (A reused 4x, B reused 2x).
// mode 0: plain (+bias). mode 1: softplus(v + bias)  (dt_proj).
__global__ void gemm_wmma_kernel(const __bf16* __restrict__ apk,
                                 const __bf16* __restrict__ bpk,
                                 const float* __restrict__ bias,
                                 float* __restrict__ out,
                                 int ksteps, int nwaven, int nout, int mode) {
    int lane = threadIdx.x & 31;
    int tile = blockIdx.x * 8 + (threadIdx.x >> 5);
    int wm = tile / nwaven;           // 32-row block index
    int wn = tile - wm * nwaven;      // 64-col block index

    const __bf16* ap[2];
    const __bf16* bp[4];
#pragma unroll
    for (int i = 0; i < 2; ++i)
        ap[i] = apk + (size_t)(2 * wm + i) * ksteps * 512 + lane * 16;
#pragma unroll
    for (int j = 0; j < 4; ++j)
        bp[j] = bpk + (size_t)(4 * wn + j) * ksteps * 512 + lane * 16;

    v8f acc[2][4] = {};
    for (int ks = 0; ks < ksteps; ++ks) {
        v16bf a[2], b[4];
#pragma unroll
        for (int i = 0; i < 2; ++i) { a[i] = gload_v16bf(ap[i]); ap[i] += 512; }
#pragma unroll
        for (int j = 0; j < 4; ++j) { b[j] = gload_v16bf(bp[j]); bp[j] += 512; }
#pragma unroll
        for (int i = 0; i < 2; ++i)
#pragma unroll
            for (int j = 0; j < 4; ++j)
                acc[i][j] = __builtin_amdgcn_wmma_f32_16x16x32_bf16(
                    false, a[i], false, b[j], (short)0, acc[i][j], false, false);
    }

    int cl = lane & 15;
    int rh = (lane >> 4) << 3;        // lanes 16-31 hold rows m+8
#pragma unroll
    for (int i = 0; i < 2; ++i) {
        int rbase = (2 * wm + i) * 16 + rh;
#pragma unroll
        for (int j = 0; j < 4; ++j) {
            int col = (4 * wn + j) * 16 + cl;
            float bv = bias ? gload_f32(&bias[col]) : 0.f;
#pragma unroll
            for (int r = 0; r < 8; ++r) {
                float v = acc[i][j][r] + bv;
                if (mode == 1) v = (v > 20.f) ? v : log1pf(__expf(v));  // softplus
                gstore_f32(&out[(size_t)(rbase + r) * nout + col], v);
            }
        }
    }
}

// --------------------- causal depthwise conv(K=3) + SiLU -------------------
__global__ void conv_silu_kernel(const float* __restrict__ hp,
                                 const float* __restrict__ cw,
                                 const float* __restrict__ cb,
                                 float* __restrict__ h,
                                 __bf16* __restrict__ hpack) {
    size_t idx = (size_t)blockIdx.x * 256 + threadIdx.x;   // M * DIN
    int d = (int)(idx & (DIN - 1));
    int m = (int)(idx >> 10);
    int l = m & (Ll - 1);
    float w0 = gload_f32(&cw[d * 3 + 0]);
    float w1 = gload_f32(&cw[d * 3 + 1]);
    float w2 = gload_f32(&cw[d * 3 + 2]);
    float v = w2 * gload_f32(&hp[idx]);
    if (l >= 1) v += w1 * gload_f32(&hp[idx - DIN]);
    if (l >= 2) v += w0 * gload_f32(&hp[idx - 2 * DIN]);
    v += gload_f32(&cb[d]);
    v = v * (1.f / (1.f + __expf(-v)));     // SiLU
    gstore_f32(&h[idx], v);
    gstore_bf16(&hpack[a_pack_idx(m, d, DIN >> 5)], (__bf16)v);
}

// --------------------------- RMSNorm split ---------------------------------
__global__ void rms_kernel(const float* __restrict__ dbc,
                           const float* __restrict__ dtln,
                           const float* __restrict__ bln,
                           const float* __restrict__ cln,
                           float* __restrict__ bm, float* __restrict__ cm,
                           __bf16* __restrict__ drms) {
    int m = blockIdx.x * 256 + threadIdx.x;   // 16384 rows
    const float* row = dbc + (size_t)m * DBC;
    float buf[DBC];
#pragma unroll
    for (int i = 0; i < DBC; ++i) buf[i] = gload_f32(&row[i]);

    float s = 0.f;
#pragma unroll
    for (int i = 0; i < DT; ++i) s += buf[i] * buf[i];
    float r = rsqrtf(s / DT + 1e-5f);
#pragma unroll
    for (int j = 0; j < DT; ++j)
        gstore_bf16(&drms[a_pack_idx(m, j, 1)], (__bf16)(buf[j] * r * gload_f32(&dtln[j])));

    s = 0.f;
#pragma unroll
    for (int i = 0; i < Ns; ++i) s += buf[DT + i] * buf[DT + i];
    r = rsqrtf(s / Ns + 1e-5f);
#pragma unroll
    for (int j = 0; j < Ns; ++j)
        gstore_f32(&bm[(size_t)m * Ns + j], buf[DT + j] * r * gload_f32(&bln[j]));

    s = 0.f;
#pragma unroll
    for (int i = 0; i < Ns; ++i) s += buf[DT + Ns + i] * buf[DT + Ns + i];
    r = rsqrtf(s / Ns + 1e-5f);
#pragma unroll
    for (int j = 0; j < Ns; ++j)
        gstore_f32(&cm[(size_t)m * Ns + j], buf[DT + Ns + j] * r * gload_f32(&cln[j]));
}

// --------------------------- selective scan --------------------------------
// thread = (b, d). 16 f32 states in registers, loop over L.
// Bm/Cm staged into LDS in 64-step chunks using async copy (gfx1250
// GLOBAL_LOAD_ASYNC_TO_LDS_B128, tracked by ASYNCcnt) when available.
__global__ void scan_kernel(const float* __restrict__ delta,
                            const float* __restrict__ h,
                            const float* __restrict__ bm,
                            const float* __restrict__ cm,
                            const float* __restrict__ aneg,
                            const float* __restrict__ dskip,
                            __bf16* __restrict__ ypack) {
    __shared__ __align__(16) float bm_s[64 * Ns];
    __shared__ __align__(16) float cm_s[64 * Ns];

    int b = blockIdx.x >> 2;                       // 4 blocks per batch
    int d = ((blockIdx.x & 3) << 8) + threadIdx.x; // 256 d per block

    float ad[Ns];
#pragma unroll
    for (int n = 0; n < Ns; ++n) ad[n] = gload_f32(&aneg[d * Ns + n]);
    float dv = gload_f32(&dskip[d]);

    float st[Ns];
#pragma unroll
    for (int n = 0; n < Ns; ++n) st[n] = 0.f;

    for (int lc = 0; lc < Ll / 64; ++lc) {
        __syncthreads();
        int base = (b * Ll + lc * 64) * Ns;        // 1024 floats per chunk
#if USE_ASYNC_LDS
        {
            int i4 = threadIdx.x * 4;              // 16 bytes per thread per array
            __builtin_amdgcn_global_load_async_to_lds_b128(
                (__attribute__((address_space(1))) v4i*)(bm + base + i4),
                (__attribute__((address_space(3))) v4i*)(&bm_s[i4]), 0, 0);
            __builtin_amdgcn_global_load_async_to_lds_b128(
                (__attribute__((address_space(1))) v4i*)(cm + base + i4),
                (__attribute__((address_space(3))) v4i*)(&cm_s[i4]), 0, 0);
            __builtin_amdgcn_s_wait_asynccnt(0);
        }
#else
        for (int i = threadIdx.x; i < 64 * Ns; i += 256) {
            bm_s[i] = gload_f32(&bm[base + i]);
            cm_s[i] = gload_f32(&cm[base + i]);
        }
#endif
        __syncthreads();

        for (int ll = 0; ll < 64; ++ll) {
            int l = lc * 64 + ll;
            size_t idx = ((size_t)(b * Ll + l) << 10) + d;
            float dt = gload_f32(&delta[idx]);
            float hv = gload_f32(&h[idx]);
            float dh = dt * hv;
            float ys = 0.f;
#pragma unroll
            for (int n = 0; n < Ns; ++n) {
                float da = __expf(dt * ad[n]);
                st[n] = fmaf(da, st[n], dh * bm_s[ll * Ns + n]);
                ys = fmaf(st[n], cm_s[ll * Ns + n], ys);
            }
            float y = ys + dv * hv;
            gstore_bf16(&ypack[a_pack_idx(b * Ll + l, d, DIN >> 5)], (__bf16)y);
        }
    }
}

// ---------------------------------------------------------------------------
extern "C" void kernel_launch(void* const* d_in, const int* in_sizes, int n_in,
                              void* d_out, int out_size, void* d_ws, size_t ws_size,
                              hipStream_t stream) {
    const float* x       = (const float*)d_in[0];   // (16,1024,512)
    const float* in_w    = (const float*)d_in[1];   // (1024,512)
    const float* in_b    = (const float*)d_in[2];   // (1024)
    const float* conv_w  = (const float*)d_in[3];   // (1024,1,3)
    const float* conv_b  = (const float*)d_in[4];   // (1024)
    const float* xproj_w = (const float*)d_in[5];   // (64,1024)
    const float* dt_w    = (const float*)d_in[6];   // (1024,32)
    const float* dt_b    = (const float*)d_in[7];   // (1024)
    const float* a_log   = (const float*)d_in[8];   // (1024,16)
    const float* d_skip  = (const float*)d_in[9];   // (1024)
    const float* out_w   = (const float*)d_in[10];  // (512,1024)
    const float* out_b   = (const float*)d_in[11];  // (512)
    const float* dtln_w  = (const float*)d_in[12];  // (32)
    const float* bln_w   = (const float*)d_in[13];  // (16)
    const float* cln_w   = (const float*)d_in[14];  // (16)
    float* out = (float*)d_out;                     // (16,1024,512)

    // ---- workspace layout (regions reused across pipeline stages) ----
    char* w = (char*)d_ws;
    size_t off = 0;
    auto take = [&](size_t bytes) -> void* {
        void* p = w + off;
        off = (off + bytes + 255) & ~(size_t)255;
        return p;
    };
    float*  h_pre  = (float*)take((size_t)Mrows * DIN * 4);  // later reused: delta
    float*  h_f32  = (float*)take((size_t)Mrows * DIN * 4);
    __bf16* hpack  = (__bf16*)take((size_t)Mrows * DIN * 2); // later reused: ypack
    char*   xreg   = (char*)take((size_t)Mrows * DM * 2);    // xpack, later dbc/Bm/Cm/drms
    __bf16* w_in   = (__bf16*)take((size_t)DIN * DM * 2);
    __bf16* w_out  = (__bf16*)take((size_t)DM * DIN * 2);
    __bf16* w_xp   = (__bf16*)take((size_t)DBC * DIN * 2);
    __bf16* w_dt   = (__bf16*)take((size_t)DIN * DT * 2);
    float*  aneg   = (float*)take((size_t)DIN * Ns * 4);

    __bf16* xpack = (__bf16*)xreg;
    float*  dbc   = (float*)xreg;                                 // 4 MB (after K1)
    float*  bm    = (float*)(xreg + (size_t)Mrows * DBC * 4);     // 1 MB
    float*  cm    = (float*)(xreg + (size_t)Mrows * DBC * 4 + (size_t)Mrows * Ns * 4);
    __bf16* drms  = (__bf16*)(xreg + (size_t)Mrows * DBC * 4 + 2 * (size_t)Mrows * Ns * 4);
    float*  delta = h_pre;    // reuse after conv consumes h_pre
    __bf16* ypack = hpack;    // reuse after x_proj consumes hpack

    // ---- stage 0: repack weights / precompute A, pack x ----
    pack_b_kernel<<<(DIN * DM) / 256, 256, 0, stream>>>(in_w, w_in, DM);
    pack_b_kernel<<<(DBC * DIN) / 256, 256, 0, stream>>>(xproj_w, w_xp, DIN);
    pack_b_kernel<<<(DIN * DT) / 256, 256, 0, stream>>>(dt_w, w_dt, DT);
    pack_b_kernel<<<(DM * DIN) / 256, 256, 0, stream>>>(out_w, w_out, DIN);
    aneg_kernel<<<(DIN * Ns) / 256, 256, 0, stream>>>(a_log, aneg);
    pack_a_kernel<<<(Mrows * DM) / 256, 256, 0, stream>>>(x, xpack, DM);

    // ---- stage 1: in_proj  h_pre[16384,1024] = x @ in_w^T + in_b ----
    gemm_wmma_kernel<<<(Mrows / 32) * (DIN / 64) / 8, 256, 0, stream>>>(
        xpack, w_in, in_b, h_pre, DM / 32, DIN / 64, DIN, 0);

    // ---- stage 2: causal conv(K=3) + SiLU -> h (f32 + packed bf16) ----
    conv_silu_kernel<<<((size_t)Mrows * DIN) / 256, 256, 0, stream>>>(
        h_pre, conv_w, conv_b, h_f32, hpack);

    // ---- stage 3: x_proj  dbc[16384,64] = h @ xproj_w^T ----
    gemm_wmma_kernel<<<(Mrows / 32) * (DBC / 64) / 8, 256, 0, stream>>>(
        hpack, w_xp, nullptr, dbc, DIN / 32, DBC / 64, DBC, 0);

    // ---- stage 4: RMSNorm split -> delta_rms(packed), Bm, Cm ----
    rms_kernel<<<Mrows / 256, 256, 0, stream>>>(dbc, dtln_w, bln_w, cln_w, bm, cm, drms);

    // ---- stage 5: dt_proj + softplus  delta[16384,1024] (reuses h_pre) ----
    gemm_wmma_kernel<<<(Mrows / 32) * (DIN / 64) / 8, 256, 0, stream>>>(
        drms, w_dt, dt_b, delta, DT / 32, DIN / 64, DIN, 1);

    // ---- stage 6: selective scan -> y (packed bf16, reuses hpack) ----
    scan_kernel<<<Bb * (DIN / 256), 256, 0, stream>>>(
        delta, h_f32, bm, cm, aneg, d_skip, ypack);

    // ---- stage 7: out_proj  out[16384,512] = y @ out_w^T + out_b ----
    gemm_wmma_kernel<<<(Mrows / 32) * (DM / 64) / 8, 256, 0, stream>>>(
        ypack, w_out, out_b, out, DIN / 32, DM / 64, DM, 0);

    (void)in_sizes; (void)n_in; (void)out_size; (void)ws_size;
}
